// Regressor_27650999452543
// MI455X (gfx1250) — matrix operand order repaired and checked by reading, not compiled, not run on hardware
//
#include <hip/hip_runtime.h>
#include <math.h>

typedef __attribute__((ext_vector_type(16))) _Float16 v16h;
typedef __attribute__((ext_vector_type(8)))  _Float16 v8h;
typedef __attribute__((ext_vector_type(8)))  float    v8f;

#define TPB 256

// ---------------------------------------------------------------------------
// Pack kernels: f32 -> zero-padded f16 (K padded to Kp, W rows padded to Np)
// ---------------------------------------------------------------------------
__global__ void pack_a_kernel(const float* __restrict__ A, _Float16* __restrict__ Ah,
                              int M, int K, int Kp, int lda)
{
    long idx = (long)blockIdx.x * blockDim.x + threadIdx.x;
    long total = (long)M * Kp;
    if (idx >= total) return;
    int k = (int)(idx % Kp); long m = idx / Kp;
    float v = (k < K) ? A[m * (size_t)lda + k] : 0.f;
    Ah[idx] = (_Float16)v;
}

__global__ void pack_w_kernel(const float* __restrict__ W, _Float16* __restrict__ Wh,
                              int N, int K, int Kp, int Np)
{
    long idx = (long)blockIdx.x * blockDim.x + threadIdx.x;
    long total = (long)Np * Kp;
    if (idx >= total) return;
    int k = (int)(idx % Kp); long n = idx / Kp;
    float v = (n < N && k < K) ? W[n * (size_t)K + k] : 0.f;
    Wh[idx] = (_Float16)v;
}

// ---------------------------------------------------------------------------
// WMMA GEMM on packed f16:  C[m,n] = act( sum_k A[m,k]*W[n,k] + bias[n] )
// Ah: M x Kp f16 (M multiple of 32), Wh: Np x Kp f16 (Np multiple of 64, rows
// >= N zero-filled).  Block = 128 thr = 4 waves (M dir); each wave owns a
// 32x64 output tile: two A fragments share four B fragments -> 8 WMMAs/step.
// ---------------------------------------------------------------------------
__global__ void gemm_wmma_packed_kernel(const _Float16* __restrict__ Ah,
                                        const _Float16* __restrict__ Wh,
                                        const float* __restrict__ bias,
                                        float* __restrict__ C,
                                        int M, int N, int Kp, int ldc, int relu)
{
    const int lane = threadIdx.x & 31;
    const int wave = threadIdx.x >> 5;
    const int tn0  = blockIdx.x << 6;              // 64 columns per block
    const int tm0  = (blockIdx.y * 4 + wave) << 5; // 32 rows per wave
    if (tm0 >= M) return;                          // wave-uniform
    const int hi = lane >> 4;
    const int lo = lane & 15;

    const _Float16* arow0 = Ah + (size_t)(tm0 + lo) * Kp;
    const _Float16* arow1 = arow0 + (size_t)16 * Kp;
    const _Float16* brow  = Wh + (size_t)(tn0 + lo) * Kp;
    const size_t bs = (size_t)16 * Kp;

    v8f acc[8] = {};
    for (int k0 = 0; k0 < Kp; k0 += 32) {
        // A fragments (16-bit A 16x32 layout): halves [8*hi..+7] and [16+8*hi..+7]
        v8h a00 = *(const v8h*)(arow0 + k0 + (hi << 3));
        v8h a01 = *(const v8h*)(arow0 + k0 + 16 + (hi << 3));
        v8h a10 = *(const v8h*)(arow1 + k0 + (hi << 3));
        v8h a11 = *(const v8h*)(arow1 + k0 + 16 + (hi << 3));
        v16h aL = __builtin_shufflevector(a00, a01, 0,1,2,3,4,5,6,7,8,9,10,11,12,13,14,15);
        v16h aH = __builtin_shufflevector(a10, a11, 0,1,2,3,4,5,6,7,8,9,10,11,12,13,14,15);
        // B fragments (16-bit B 32x16 layout): halves [8*hi..+7] and [16+8*hi..+7]
        v16h b0 = __builtin_shufflevector(*(const v8h*)(brow        + k0 + (hi << 3)), *(const v8h*)(brow        + k0 + 16 + (hi << 3)), 0,1,2,3,4,5,6,7,8,9,10,11,12,13,14,15);
        v16h b1 = __builtin_shufflevector(*(const v8h*)(brow + bs   + k0 + (hi << 3)), *(const v8h*)(brow + bs   + k0 + 16 + (hi << 3)), 0,1,2,3,4,5,6,7,8,9,10,11,12,13,14,15);
        v16h b2 = __builtin_shufflevector(*(const v8h*)(brow + bs*2 + k0 + (hi << 3)), *(const v8h*)(brow + bs*2 + k0 + 16 + (hi << 3)), 0,1,2,3,4,5,6,7,8,9,10,11,12,13,14,15);
        v16h b3 = __builtin_shufflevector(*(const v8h*)(brow + bs*3 + k0 + (hi << 3)), *(const v8h*)(brow + bs*3 + k0 + 16 + (hi << 3)), 0,1,2,3,4,5,6,7,8,9,10,11,12,13,14,15);
        acc[0] = __builtin_amdgcn_wmma_f32_16x16x32_f16(false, aL, false, b0, (short)0, acc[0], false, false);
        acc[1] = __builtin_amdgcn_wmma_f32_16x16x32_f16(false, aL, false, b1, (short)0, acc[1], false, false);
        acc[2] = __builtin_amdgcn_wmma_f32_16x16x32_f16(false, aL, false, b2, (short)0, acc[2], false, false);
        acc[3] = __builtin_amdgcn_wmma_f32_16x16x32_f16(false, aL, false, b3, (short)0, acc[3], false, false);
        acc[4] = __builtin_amdgcn_wmma_f32_16x16x32_f16(false, aH, false, b0, (short)0, acc[4], false, false);
        acc[5] = __builtin_amdgcn_wmma_f32_16x16x32_f16(false, aH, false, b1, (short)0, acc[5], false, false);
        acc[6] = __builtin_amdgcn_wmma_f32_16x16x32_f16(false, aH, false, b2, (short)0, acc[6], false, false);
        acc[7] = __builtin_amdgcn_wmma_f32_16x16x32_f16(false, aH, false, b3, (short)0, acc[7], false, false);
    }

    // store: C/D layout -> row m = mbase + 8*hi + r, col n = tn0 + 16*(t&3) + lo
#pragma unroll
    for (int t = 0; t < 8; ++t) {
        int bn = tn0 + ((t & 3) << 4) + lo;
        int mbase = tm0 + ((t >> 2) << 4);
        if (bn < N) {
            float bv = bias ? bias[bn] : 0.f;
#pragma unroll
            for (int r = 0; r < 8; ++r) {
                float v = acc[t][r] + bv;
                if (relu) v = fmaxf(v, 0.f);
                C[(size_t)(mbase + (hi << 3) + r) * ldc + bn] = v;  // M % 32 == 0
            }
        }
    }
}

// ---------------------------------------------------------------------------
// Direct 3x3x3 conv, padding 1 (NCDHW)
// ---------------------------------------------------------------------------
__global__ void conv3d_k3p1_kernel(const float* __restrict__ x, const float* __restrict__ w,
                                   const float* __restrict__ bias, float* __restrict__ y,
                                   int Mb, int Cin, int Cout, int D, int H, int Wd)
{
    long idx = (long)blockIdx.x * blockDim.x + threadIdx.x;
    long total = (long)Mb * Cout * D * H * Wd;
    if (idx >= total) return;
    int wi = idx % Wd; long t = idx / Wd;
    int hi = t % H;    t /= H;
    int di = t % D;    t /= D;
    int co = t % Cout; int m = (int)(t / Cout);

    float acc = bias[co];
    const float* xb = x + (long)m * Cin * D * H * Wd;
    const float* wb = w + (long)co * Cin * 27;
    for (int ci = 0; ci < Cin; ++ci) {
        const float* xc = xb + (long)ci * D * H * Wd;
        const float* wc = wb + ci * 27;
        for (int kd = 0; kd < 3; ++kd) {
            int dd = di + kd - 1; if (dd < 0 || dd >= D) continue;
            for (int kh = 0; kh < 3; ++kh) {
                int hh = hi + kh - 1; if (hh < 0 || hh >= H) continue;
                for (int kw = 0; kw < 3; ++kw) {
                    int ww = wi + kw - 1; if (ww < 0 || ww >= Wd) continue;
                    acc += xc[(dd * H + hh) * Wd + ww] * wc[(kd * 3 + kh) * 3 + kw];
                }
            }
        }
    }
    y[idx] = acc;
}

// ---------------------------------------------------------------------------
// 2x2x2 max-pool, stride 2
// ---------------------------------------------------------------------------
__global__ void maxpool3d_kernel(const float* __restrict__ x, float* __restrict__ y,
                                 long MC, int D, int H, int Wd,
                                 int pd, int ph, int pw, int Do, int Ho, int Wo)
{
    long idx = (long)blockIdx.x * blockDim.x + threadIdx.x;
    long total = MC * Do * Ho * Wo;
    if (idx >= total) return;
    int wo = idx % Wo; long t = idx / Wo;
    int ho = t % Ho;   t /= Ho;
    int dO = t % Do;   long mc = t / Do;
    const float* xb = x + mc * (long)D * H * Wd;
    float m = -INFINITY;
    for (int kd = 0; kd < 2; ++kd) {
        int dd = dO * 2 - pd + kd; if (dd < 0 || dd >= D) continue;
        for (int kh = 0; kh < 2; ++kh) {
            int hh = ho * 2 - ph + kh; if (hh < 0 || hh >= H) continue;
            for (int kw = 0; kw < 2; ++kw) {
                int ww = wo * 2 - pw + kw; if (ww < 0 || ww >= Wd) continue;
                m = fmaxf(m, xb[(dd * H + hh) * Wd + ww]);
            }
        }
    }
    y[idx] = m;
}

// ---------------------------------------------------------------------------
// BatchNorm (training mode, biased variance) statistics + apply
// ---------------------------------------------------------------------------
__global__ void bn_stats_conv_kernel(const float* __restrict__ x, float* mean, float* var,
                                     int Mb, int C, int S)
{
    int c = blockIdx.x;
    __shared__ float ss[TPB], sq[TPB];
    float s1 = 0.f, s2 = 0.f;
    long n = (long)Mb * S;
    for (long i = threadIdx.x; i < n; i += TPB) {
        long m = i / S, sp = i % S;
        float v = x[((m * C) + c) * S + sp];
        s1 += v; s2 += v * v;
    }
    ss[threadIdx.x] = s1; sq[threadIdx.x] = s2; __syncthreads();
    for (int o = TPB / 2; o > 0; o >>= 1) {
        if ((int)threadIdx.x < o) { ss[threadIdx.x] += ss[threadIdx.x + o]; sq[threadIdx.x] += sq[threadIdx.x + o]; }
        __syncthreads();
    }
    if (threadIdx.x == 0) {
        float mu = ss[0] / (float)n;
        mean[c] = mu; var[c] = sq[0] / (float)n - mu * mu;
    }
}

__global__ void bn_apply_conv_kernel(float* x, const float* __restrict__ mean,
                                     const float* __restrict__ var,
                                     const float* __restrict__ g, const float* __restrict__ b,
                                     int C, int S, long total, int relu)
{
    long idx = (long)blockIdx.x * blockDim.x + threadIdx.x;
    if (idx >= total) return;
    int c = (int)((idx / S) % C);
    float v = (x[idx] - mean[c]) * rsqrtf(var[c] + 1e-5f) * g[c] + b[c];
    if (relu) v = fmaxf(v, 0.f);
    x[idx] = v;
}

__global__ void bn_stats_fc_kernel(const float* __restrict__ x, float* mean, float* var,
                                   int Rows, int Cols)
{
    int j = blockIdx.x;
    __shared__ float ss[TPB], sq[TPB];
    float s1 = 0.f, s2 = 0.f;
    for (int i = threadIdx.x; i < Rows; i += TPB) {
        float v = x[(size_t)i * Cols + j];
        s1 += v; s2 += v * v;
    }
    ss[threadIdx.x] = s1; sq[threadIdx.x] = s2; __syncthreads();
    for (int o = TPB / 2; o > 0; o >>= 1) {
        if ((int)threadIdx.x < o) { ss[threadIdx.x] += ss[threadIdx.x + o]; sq[threadIdx.x] += sq[threadIdx.x + o]; }
        __syncthreads();
    }
    if (threadIdx.x == 0) {
        float mu = ss[0] / (float)Rows;
        mean[j] = mu; var[j] = sq[0] / (float)Rows - mu * mu;
    }
}

__global__ void bn_apply_fc_kernel(float* x, const float* __restrict__ mean,
                                   const float* __restrict__ var,
                                   const float* __restrict__ g, const float* __restrict__ b,
                                   int Cols, long total, int relu)
{
    long idx = (long)blockIdx.x * blockDim.x + threadIdx.x;
    if (idx >= total) return;
    int c = (int)(idx % Cols);
    float v = (x[idx] - mean[c]) * rsqrtf(var[c] + 1e-5f) * g[c] + b[c];
    if (relu) v = fmaxf(v, 0.f);
    x[idx] = v;
}

// ---------------------------------------------------------------------------
// GRU gate fusion (torch order: r,z,n)
// ---------------------------------------------------------------------------
__global__ void gru_fuse_kernel(const float* __restrict__ gi, const float* __restrict__ gh,
                                float* __restrict__ h, float* __restrict__ seqout, int t, int T)
{
    int idx = blockIdx.x * blockDim.x + threadIdx.x;
    if (idx >= 64 * 256) return;
    int bc = idx >> 8, j = idx & 255;
    const float* gir = gi + (size_t)bc * 768;
    const float* ghr = gh + (size_t)bc * 768;
    float r = 1.f / (1.f + expf(-(gir[j] + ghr[j])));
    float z = 1.f / (1.f + expf(-(gir[256 + j] + ghr[256 + j])));
    float n = tanhf(gir[512 + j] + r * ghr[512 + j]);
    float hv = (1.f - z) * n + z * h[idx];
    h[idx] = hv;
    seqout[((size_t)bc * T + t) * 256 + j] = hv;
}

// ---------------------------------------------------------------------------
// GNN helpers
// ---------------------------------------------------------------------------
__global__ void fill_kernel(float* p, float v, long n)
{
    long i = (long)blockIdx.x * blockDim.x + threadIdx.x;
    if (i < n) p[i] = v;
}

__global__ void feat_kernel(const float* __restrict__ node_x, const float* __restrict__ enc,
                            const int* __restrict__ cell, float* __restrict__ feat)
{
    long idx = (long)blockIdx.x * blockDim.x + threadIdx.x;
    const long total = 2L * 10000 * 515;
    if (idx >= total) return;
    int f = (int)(idx % 515); long t = idx / 515;
    int n = (int)(t % 10000); int b = (int)(t / 10000);
    float v;
    if (f < 3) v = node_x[n * 3 + f];
    else       v = enc[((size_t)(b * 32 + cell[n])) * 512 + (f - 3)];
    feat[idx] = v;
}

__global__ void edge_build_kernel(const int* __restrict__ edge_index,
                                  int* __restrict__ src, int* __restrict__ dst, float* cnt,
                                  int Eb, int Et)
{
    int i = blockIdx.x * blockDim.x + threadIdx.x;
    if (i >= Et) return;
    int s, d;
    if (i < Eb) { s = edge_index[i]; d = edge_index[Eb + i]; }
    else        { s = d = i - Eb; }                 // self loops
    src[i] = s; dst[i] = d;
    atomicAdd(&cnt[d], 1.0f);
}

__device__ inline void atomicMaxFloat(float* addr, float val)
{
    if (val >= 0.f) atomicMax((int*)addr, __float_as_int(val));
    else            atomicMin((unsigned int*)addr, __float_as_uint(val));
}

__global__ void gat_scores_kernel(const int* __restrict__ src, const int* __restrict__ dst,
                                  const float* __restrict__ xl, const float* __restrict__ xr,
                                  const float* __restrict__ att, float* __restrict__ e,
                                  float* emax, int Et, int h, int fo)
{
    int t = blockIdx.x * blockDim.x + threadIdx.x;
    if (t >= Et * h) return;
    int eidx = t / h, hh = t % h;
    int s = src[eidx], d = dst[eidx];
    const float* pl = xl + ((size_t)s * h + hh) * fo;
    const float* pr = xr + ((size_t)d * h + hh) * fo;
    const float* pa = att + (size_t)hh * fo;
    float sum = 0.f;
    for (int f = 0; f < fo; ++f) {
        float v = pl[f] + pr[f];
        v = v > 0.f ? v : 0.2f * v;                 // leaky_relu(., 0.2)
        sum += v * pa[f];
    }
    e[t] = sum;
    atomicMaxFloat(&emax[(size_t)d * h + hh], sum);
}

__global__ void gat_exp_kernel(const int* __restrict__ dst, float* __restrict__ e,
                               const float* __restrict__ emax, float* den, int Et, int h)
{
    int t = blockIdx.x * blockDim.x + threadIdx.x;
    if (t >= Et * h) return;
    int eidx = t / h, hh = t % h;
    int d = dst[eidx];
    float ee = expf(e[t] - emax[(size_t)d * h + hh]);
    e[t] = ee;
    atomicAdd(&den[(size_t)d * h + hh], ee);
}

__global__ void gat_agg_kernel(const int* __restrict__ src, const int* __restrict__ dst,
                               const float* __restrict__ ee, const float* __restrict__ den,
                               const float* __restrict__ xl, float* agg, long total, int h, int fo)
{
    long t = (long)blockIdx.x * blockDim.x + threadIdx.x;
    if (t >= total) return;
    int f = (int)(t % fo); long u = t / fo;
    int hh = (int)(u % h); int eidx = (int)(u / h);
    int s = src[eidx], d = dst[eidx];
    float alpha = ee[(size_t)eidx * h + hh] / den[(size_t)d * h + hh];
    atomicAdd(&agg[((size_t)d * h + hh) * fo + f],
              alpha * xl[((size_t)s * h + hh) * fo + f]);
}

__global__ void gat_final_kernel(const float* __restrict__ agg, const float* __restrict__ cnt,
                                 const float* __restrict__ bias, float* __restrict__ out,
                                 int NN, int hfo)
{
    long idx = (long)blockIdx.x * blockDim.x + threadIdx.x;
    if (idx >= (long)NN * hfo) return;
    int n = (int)(idx / hfo), j = (int)(idx % hfo);
    out[idx] = agg[idx] / cnt[n] + bias[j];
}

__global__ void output_kernel(const float* __restrict__ x, const int* __restrict__ mask,
                              const float* __restrict__ y, float* __restrict__ out, int NN)
{
    int i = blockIdx.x * blockDim.x + threadIdx.x;
    if (i >= NN) return;
    bool mk = mask[i] != 0;
    out[i]      = mk ? expf(x[i]) : 0.f;
    out[NN + i] = mk ? y[i] : 0.f;
}

// ---------------------------------------------------------------------------
// Host orchestration
// ---------------------------------------------------------------------------
extern "C" void kernel_launch(void* const* d_in, const int* in_sizes, int n_in,
                              void* d_out, int out_size, void* d_ws, size_t ws_size,
                              hipStream_t stream)
{
    (void)in_sizes; (void)n_in; (void)out_size; (void)ws_size;
    const int M3 = 1600;          // B*NC*T
    const int NN = 20000;         // B*N
    const int EB = 240000;        // B*E
    const int ET = 260000;        // + self loops

    // -------- workspace layout (32B-aligned slots) --------
    float* wsf = (float*)d_ws;
    size_t off = 0;
    auto alloc = [&](size_t n) { float* p = wsf + off; off += (n + 7) & ~(size_t)7; return p; };
    float* BIG0 = alloc(19660800);
    float* BIG1 = alloc(19660800);
    float* SEQ0 = alloc(409600);
    float* SEQ1 = alloc(409600);
    float* GI   = alloc(49152);
    float* GH   = alloc(49152);
    float* HB   = alloc(16384);
    float* ENCB = alloc(32768);
    float* EARR = alloc(520000);
    float* EMAX = alloc(40000);
    float* DEN  = alloc(40000);
    float* CNT  = alloc(20000);
    int*   SRC  = (int*)alloc(260000);
    int*   DST  = (int*)alloc(260000);
    float* MEAN = alloc(1024);
    float* VAR  = alloc(1024);
    _Float16* AH = (_Float16*)alloc(5440000);   // 10.88M halfs (20000 x 544 max)
    _Float16* WH = (_Float16*)alloc(2097152);   // 4.19M halfs (512 x 6400 + pad max)
    // GNN overlays on the (dead by then) conv buffers
    float* X0 = BIG0;                 // up to 20000 x 515
    float* X1 = BIG0 + 10300000;      // up to 20000 x 256
    float* XL = BIG1;                 // 20000 x 256
    float* XR = BIG1 + 5120000;
    float* AGG = BIG1 + 10240000;

    auto GB = [](long n) { return (unsigned)((n + TPB - 1) / TPB); };
    auto gemm = [&](const float* A, int lda, const float* W, const float* bias, float* C,
                    int M, int N, int K, int ldc, int relu) {
        int Kp = (K + 31) & ~31;
        int Np = (N + 63) & ~63;
        pack_a_kernel<<<GB((long)M * Kp), TPB, 0, stream>>>(A, AH, M, K, Kp, lda);
        pack_w_kernel<<<GB((long)Np * Kp), TPB, 0, stream>>>(W, WH, N, K, Kp, Np);
        dim3 grid(Np / 64, (M + 127) / 128);
        gemm_wmma_packed_kernel<<<grid, 128, 0, stream>>>(AH, WH, bias, C, M, N, Kp, ldc, relu);
    };
    auto bn_conv = [&](float* x, int Mb, int C, int S, const float* g, const float* b, int relu) {
        bn_stats_conv_kernel<<<C, TPB, 0, stream>>>(x, MEAN, VAR, Mb, C, S);
        long tot = (long)Mb * C * S;
        bn_apply_conv_kernel<<<GB(tot), TPB, 0, stream>>>(x, MEAN, VAR, g, b, C, S, tot, relu);
    };
    auto bn_fc = [&](float* x, int Rows, int Cols, const float* g, const float* b, int relu) {
        bn_stats_fc_kernel<<<Cols, TPB, 0, stream>>>(x, MEAN, VAR, Rows, Cols);
        long tot = (long)Rows * Cols;
        bn_apply_fc_kernel<<<GB(tot), TPB, 0, stream>>>(x, MEAN, VAR, g, b, Cols, tot, relu);
    };
    auto F = [&](int i) { return (const float*)d_in[i]; };

    // ---------------- encoder ----------------
    conv3d_k3p1_kernel<<<GB((long)M3 * 64 * 180), TPB, 0, stream>>>(F(0), F(6), F(7), BIG0, M3, 5, 64, 5, 6, 6);
    bn_conv(BIG0, M3, 64, 180, F(8), F(9), 1);
    conv3d_k3p1_kernel<<<GB((long)M3 * 64 * 180), TPB, 0, stream>>>(BIG0, F(10), F(11), BIG1, M3, 64, 64, 5, 6, 6);
    bn_conv(BIG1, M3, 64, 180, F(12), F(13), 1);
    maxpool3d_kernel<<<GB((long)M3 * 64 * 48), TPB, 0, stream>>>(BIG1, BIG0, (long)M3 * 64, 5, 6, 6, 1, 1, 1, 3, 4, 4);
    conv3d_k3p1_kernel<<<GB((long)M3 * 256 * 48), TPB, 0, stream>>>(BIG0, F(14), F(15), BIG1, M3, 64, 256, 3, 4, 4);
    bn_conv(BIG1, M3, 256, 48, F(16), F(17), 1);
    maxpool3d_kernel<<<GB((long)M3 * 256 * 8), TPB, 0, stream>>>(BIG1, BIG0, (long)M3 * 256, 3, 4, 4, 1, 0, 0, 2, 2, 2);
    // l1: [1600,2048]@[512,2048]^T -> BIG1, bn+relu
    gemm(BIG0, 2048, F(18), F(19), BIG1, M3, 512, 2048, 512, 0);
    bn_fc(BIG1, M3, 512, F(20), F(21), 1);
    // l2: -> SEQ0 [1600,256] == [64,25,256]
    gemm(BIG1, 512, F(22), F(23), SEQ0, M3, 256, 512, 256, 0);
    bn_fc(SEQ0, M3, 256, F(24), F(25), 1);

    // ---------------- GRU (2 layers, T=25) ----------------
    {
        const float* seqin = SEQ0; float* seqout = SEQ1;
        for (int l = 0; l < 2; ++l) {
            int pb = 26 + l * 4;
            fill_kernel<<<GB(64 * 256), TPB, 0, stream>>>(HB, 0.f, 64 * 256);
            for (int t = 0; t < 25; ++t) {
                gemm(seqin + (size_t)t * 256, 25 * 256, F(pb + 0), F(pb + 2), GI, 64, 768, 256, 768, 0);
                gemm(HB, 256,                F(pb + 1), F(pb + 3), GH, 64, 768, 256, 768, 0);
                gru_fuse_kernel<<<GB(64 * 256), TPB, 0, stream>>>(GI, GH, HB, seqout, t, 25);
            }
            seqin = seqout; seqout = (l == 0) ? SEQ0 : SEQ1;
        }
    }
    // dense: [64,6400]@[512,6400]^T -> ENCB [64,512], relu
    gemm(SEQ0, 6400, F(34), F(35), ENCB, 64, 512, 6400, 512, 1);

    // ---------------- feature assembly + gbn0 ----------------
    feat_kernel<<<GB((long)NN * 515), TPB, 0, stream>>>(F(1), ENCB, (const int*)d_in[2], X0);
    bn_fc(X0, NN, 515, F(64), F(65), 0);

    // ---------------- edges (+ self loops) ----------------
    fill_kernel<<<GB(NN), TPB, 0, stream>>>(CNT, 0.f, NN);
    edge_build_kernel<<<GB(ET), TPB, 0, stream>>>((const int*)d_in[3], SRC, DST, CNT, EB, ET);

    // ---------------- 7 GATv2 layers ----------------
    const int fi_a[7] = {515, 256, 256, 128, 128, 128, 128};
    const int fo_a[7] = {128, 128, 128, 128, 128, 128, 1};
    const int h_a[7]  = {2, 2, 1, 1, 1, 1, 1};
    float* xcur = X0; float* xnext = X1;
    for (int i = 0; i < 7; ++i) {
        int fi = fi_a[i], fo = fo_a[i], h = h_a[i], hfo = h * fo;
        int pb = 36 + i * 4;
        gemm(xcur, fi, F(pb + 0), nullptr, XL, NN, hfo, fi, hfo, 0);   // xl = x @ wl^T
        gemm(xcur, fi, F(pb + 1), nullptr, XR, NN, hfo, fi, hfo, 0);   // xr = x @ wr^T
        fill_kernel<<<GB((long)NN * h), TPB, 0, stream>>>(EMAX, -INFINITY, (long)NN * h);
        fill_kernel<<<GB((long)NN * h), TPB, 0, stream>>>(DEN, 0.f, (long)NN * h);
        fill_kernel<<<GB((long)NN * hfo), TPB, 0, stream>>>(AGG, 0.f, (long)NN * hfo);
        gat_scores_kernel<<<GB((long)ET * h), TPB, 0, stream>>>(SRC, DST, XL, XR, F(pb + 2), EARR, EMAX, ET, h, fo);
        gat_exp_kernel<<<GB((long)ET * h), TPB, 0, stream>>>(DST, EARR, EMAX, DEN, ET, h);
        gat_agg_kernel<<<GB((long)ET * h * fo), TPB, 0, stream>>>(SRC, DST, EARR, DEN, XL, AGG, (long)ET * h * fo, h, fo);
        gat_final_kernel<<<GB((long)NN * hfo), TPB, 0, stream>>>(AGG, CNT, F(pb + 3), xnext, NN, hfo);
        if (i < 6) bn_fc(xnext, NN, hfo, F(64 + 2 * (i + 1)), F(65 + 2 * (i + 1)), 1);
        float* tmp = xcur; xcur = xnext; xnext = tmp;
    }

    // ---------------- outputs ----------------
    output_kernel<<<GB(NN), TPB, 0, stream>>>(xcur, (const int*)d_in[4], F(5), (float*)d_out, NN);
}
